// FeatureNet_43851616092543
// MI455X (gfx1250) — matrix operand order repaired and checked by reading, not compiled, not run on hardware
//
#include <hip/hip_runtime.h>
#include <cstdint>

#define B_      2
#define T_      4096
#define M_      (B_*T_)
#define DIN_    128
#define DFEAT_  768
#define NLAYERS_ 2
#define DSTATE_ 16
#define DCONV_  4
#define DINNER_ 1536
#define DTRANK_ 48
#define DBLW_   (DTRANK_ + 2*DSTATE_)   // 80

typedef __bf16 bf16_t;
typedef bf16_t v16bf __attribute__((ext_vector_type(16)));
typedef float  v8f   __attribute__((ext_vector_type(8)));

// pointee type expected by the async-LDS builtins (from hipcc diagnostic):
// 'int __attribute__((vector_size(4*sizeof(int))))'
typedef int v4i_async __attribute__((vector_size(4 * sizeof(int))));

union Frag { uint4 q[2]; v16bf v; };

// CDNA5 async global->LDS path (ASYNCcnt-tracked). Guarded: falls back to
// synchronous LDS staging if this toolchain lacks the builtins (and on the
// host pass, where amdgcn builtins don't exist).
#if __has_builtin(__builtin_amdgcn_global_load_async_to_lds_b128) && \
    __has_builtin(__builtin_amdgcn_s_wait_asynccnt)
#define HAVE_ASYNC_LDS 1
#else
#define HAVE_ASYNC_LDS 0
#endif

__device__ __forceinline__ unsigned short f2bf(float f) {
  unsigned int x = __float_as_uint(f);
  unsigned int r = x + 0x7FFFu + ((x >> 16) & 1u);     // round-to-nearest-even
  if ((x & 0x7F800000u) == 0x7F800000u) r = x;          // inf/nan: truncate
  return (unsigned short)(r >> 16);
}

// Stage one 32-K-deep B slab (NTPW*16 columns of W, 64B each) into LDS.
// Thread j moves 16 bytes: col = j/4, 16B chunk = j%4  ->  lds byte j*16.
template<int NTPW>
__device__ __forceinline__ void stage_B(const unsigned short* __restrict__ gbase,
                                        unsigned short* lbase, int Kr, int tid)
{
  if (tid < NTPW * 64) {
    const unsigned short* gp = gbase + (size_t)(tid >> 2) * Kr + (tid & 3) * 8;
    unsigned short* lp = lbase + tid * 8;
#if HAVE_ASYNC_LDS
    __builtin_amdgcn_global_load_async_to_lds_b128(
        (__attribute__((address_space(1))) v4i_async*)gp,
        (__attribute__((address_space(3))) v4i_async*)lp, 0, 0);
#else
    *(uint4*)lp = *(const uint4*)gp;
#endif
  }
}

__device__ __forceinline__ void publish_lds()
{
#if HAVE_ASYNC_LDS
  __builtin_amdgcn_s_wait_asynccnt(0);   // own wave's async stores complete
#endif
  __syncthreads();                        // all waves' parts visible
}

// ---------------------------------------------------------------------------
// bf16 WMMA GEMM: D[M,N] = act(A[M,K] @ W[N,K]^T + bias), f32 accumulate.
// Block (8 waves) owns a 128x(NTPW*16) macro-tile: wave i -> M-tile i, all
// waves share one B slab staged to LDS (double-buffered, async DMA overlaps
// the WMMAs). NTPW is compile-time so the WMMA loop is straight-line code
// with EXEC all-ones (ISA 7.12 requirement) and no saveexec churn.
// K multiple of 32; M multiple of 128.
// act: 0=none, 1=softplus.  accum: 0=store, 1=D += result.
// ---------------------------------------------------------------------------
template<int NTPW>
__global__ void gemm_bf16_wmma(const unsigned short* __restrict__ A,
                               const unsigned short* __restrict__ W,
                               const float* __restrict__ bias,
                               float* __restrict__ D,
                               int Mr, int Nr, int Kr,
                               int ntile0, int nblocks_n, int act, int accum)
{
  __shared__ unsigned short ldsB[2][NTPW * 512];   // 2 x (NTPW*16 cols x 64B)

  int tid  = threadIdx.x;
  int wid  = tid >> 5;
  int lane = tid & 31;
  int half = lane >> 4;
  int r    = lane & 15;

  int bn    = blockIdx.x % nblocks_n;
  int bm    = blockIdx.x / nblocks_n;
  int mtile = bm * 8 + wid;
  int col0  = (ntile0 + bn * NTPW) * 16;

  const unsigned short* Arow = A + (size_t)(mtile * 16 + r) * Kr;
  const unsigned short* Wblk = W + (size_t)col0 * Kr;

  v8f acc[NTPW] = {};

  stage_B<NTPW>(Wblk, &ldsB[0][0], Kr, tid);       // prologue: k0 = 0
  publish_lds();

  int nsteps = Kr >> 5;
  for (int it = 0; it < nsteps; ++it) {
    int cur = it & 1;
    int k0  = it << 5;
    if (it + 1 < nsteps)                           // DMA next slab now,
      stage_B<NTPW>(Wblk + k0 + 32, &ldsB[1 - cur][0], Kr, tid);   // overlap WMMAs

    // A fragment (ISA 16-bit A 16x32 layout): lane = row r,
    // K = k0+8*half+[0..7] and k0+16+8*half+[0..7] -> two 16B loads.
    Frag af;
    const unsigned short* pa = Arow + k0 + 8 * half;
    af.q[0] = *(const uint4*)(pa);
    af.q[1] = *(const uint4*)(pa + 16);
    __builtin_prefetch(pa + 32, 0, 3);             // next K-step A

    #pragma unroll
    for (int i = 0; i < NTPW; ++i) {
      // B fragment (32x16): lane = column i*16+r, K = 8*half+[0..7] and 16+8*half+[0..7].
      const unsigned short* lb = &ldsB[cur][(i * 16 + r) * 32 + half * 8];
      Frag wf;
      wf.q[0] = *(const uint4*)(lb);
      wf.q[1] = *(const uint4*)(lb + 16);
      acc[i] = __builtin_amdgcn_wmma_f32_16x16x32_bf16(
                   false, af.v, false, wf.v, (short)0, acc[i], false, false);
    }
    publish_lds();   // staging of next slab done + all reads of current done
  }

  // Epilogue. C/D layout: acc[i][v] -> (row = mtile*16 + v + 8*half, col)
  #pragma unroll
  for (int i = 0; i < NTPW; ++i) {
    int col = col0 + i * 16 + r;
    float bv = bias ? bias[col] : 0.0f;
    #pragma unroll
    for (int v = 0; v < 8; ++v) {
      int row = mtile * 16 + v + 8 * half;
      float x = acc[i][v] + bv;
      if (act == 1) x = (x > 20.0f) ? x : log1pf(__expf(x));   // softplus
      size_t idx = (size_t)row * Nr + col;
      if (accum) D[idx] += x; else D[idx] = x;
    }
  }
}

// ---------------------------------------------------------------------------
// f32 -> bf16 staging with optional column extract + zero pad.
// dst[r, c(0..Cp)] = c < C ? bf16(src[r*S + off + c]) : 0
// ---------------------------------------------------------------------------
__global__ void cvt_pad_bf16(const float* __restrict__ src,
                             unsigned short* __restrict__ dst,
                             int C, int Cp, int S, int off, long long total)
{
  long long i = (long long)blockIdx.x * blockDim.x + threadIdx.x;
  if (i >= total) return;
  int c = (int)(i % Cp);
  long long r = i / Cp;
  float v = (c < C) ? src[r * (long long)S + off + c] : 0.0f;
  dst[i] = f2bf(v);
}

// ---------------------------------------------------------------------------
// Depthwise causal conv (width 4, left pad 3) + bias + SiLU.
// ---------------------------------------------------------------------------
__global__ void conv_silu(const float* __restrict__ xz,
                          const float* __restrict__ cw,
                          const float* __restrict__ cb,
                          float* __restrict__ u,
                          unsigned short* __restrict__ ubf)
{
  long long i = (long long)blockIdx.x * blockDim.x + threadIdx.x;
  if (i >= (long long)M_ * DINNER_) return;
  int d = (int)(i % DINNER_);
  int m = (int)(i / DINNER_);
  int t = m % T_;
  int b = m / T_;
  float acc = cb[d];
  #pragma unroll
  for (int j = 0; j < DCONV_; ++j) {
    int tt = t - (DCONV_ - 1) + j;
    if (tt >= 0)
      acc += cw[d * DCONV_ + j] * xz[(size_t)(b * T_ + tt) * (2 * DINNER_) + d];
  }
  float s = acc / (1.0f + __expf(-acc));   // silu
  u[i]   = s;
  ubf[i] = f2bf(s);
}

// ---------------------------------------------------------------------------
// Selective scan: one lane per (batch, channel); 16-state recurrence in VGPRs.
// Fuses D skip-connection, SiLU gating, and bf16 downconvert for out_proj.
// ---------------------------------------------------------------------------
__global__ void mamba_scan(const float* __restrict__ delta,
                           const float* __restrict__ u,
                           const float* __restrict__ dbl,
                           const float* __restrict__ xz,      // gate half
                           const float* __restrict__ Alog,
                           const float* __restrict__ Dv,
                           unsigned short* __restrict__ pbf)
{
  int tid = blockIdx.x * blockDim.x + threadIdx.x;
  if (tid >= B_ * DINNER_) return;
  int b = tid / DINNER_;
  int d = tid % DINNER_;

  float Arow[DSTATE_], h[DSTATE_];
  #pragma unroll
  for (int s = 0; s < DSTATE_; ++s) {
    Arow[s] = -__expf(Alog[d * DSTATE_ + s]);
    h[s] = 0.0f;
  }
  float Dd = Dv[d];

  for (int t = 0; t < T_; ++t) {
    size_t m = (size_t)b * T_ + t;
    float dt = delta[m * DINNER_ + d];
    float ut = u[m * DINNER_ + d];
    const float* bc = dbl + m * DBLW_;
    float du = dt * ut;
    float y = 0.0f;
    #pragma unroll
    for (int s = 0; s < DSTATE_; ++s) {
      float dA = __expf(dt * Arow[s]);
      h[s] = h[s] * dA + du * bc[DTRANK_ + s];            // B_t
      y   += h[s] * bc[DTRANK_ + DSTATE_ + s];            // C_t
    }
    float yo = y + ut * Dd;
    float g  = xz[m * (2 * DINNER_) + DINNER_ + d];       // gate
    float p  = yo * (g / (1.0f + __expf(-g)));            // y * silu(gate)
    pbf[m * DINNER_ + d] = f2bf(p);
  }
}

// ---------------------------------------------------------------------------
// LayerNorm: one wave32 per 768-wide row, shuffle-tree reductions.
// ---------------------------------------------------------------------------
__global__ void layernorm(const float* __restrict__ z,
                          const float* __restrict__ g,
                          const float* __restrict__ bta,
                          float* __restrict__ out)
{
  int wave = (blockIdx.x * blockDim.x + threadIdx.x) >> 5;
  int lane = threadIdx.x & 31;
  if (wave >= M_) return;
  const float* row = z + (size_t)wave * DFEAT_;
  float s = 0.0f;
  for (int c = lane; c < DFEAT_; c += 32) s += row[c];
  #pragma unroll
  for (int o = 16; o > 0; o >>= 1) s += __shfl_xor(s, o, 32);
  float mu = s / DFEAT_;
  float v = 0.0f;
  for (int c = lane; c < DFEAT_; c += 32) { float dd = row[c] - mu; v += dd * dd; }
  #pragma unroll
  for (int o = 16; o > 0; o >>= 1) v += __shfl_xor(v, o, 32);
  float rstd = rsqrtf(v / DFEAT_ + 1e-5f);
  float* orow = out + (size_t)wave * DFEAT_;
  for (int c = lane; c < DFEAT_; c += 32)
    orow[c] = (row[c] - mu) * rstd * g[c] + bta[c];
}

// ---------------------------------------------------------------------------

static inline int cdiv_ll(long long a, long long b) { return (int)((a + b - 1) / b); }

extern "C" void kernel_launch(void* const* d_in, const int* in_sizes, int n_in,
                              void* d_out, int out_size, void* d_ws, size_t ws_size,
                              hipStream_t stream)
{
  const float* x    = (const float*)d_in[0];
  const float* piw  = (const float*)d_in[1];   // (768,128)
  const float* pib  = (const float*)d_in[2];   // (768)
  const float* inw  = (const float*)d_in[3];   // (L,3072,768)
  const float* cw   = (const float*)d_in[4];   // (L,1536,1,4)
  const float* cb   = (const float*)d_in[5];   // (L,1536)
  const float* xpw  = (const float*)d_in[6];   // (L,80,1536)
  const float* dtw  = (const float*)d_in[7];   // (L,1536,48)
  const float* dtb  = (const float*)d_in[8];   // (L,1536)
  const float* alog = (const float*)d_in[9];   // (L,1536,16)
  const float* Dvp  = (const float*)d_in[10];  // (L,1536)
  const float* ow   = (const float*)d_in[11];  // (L,768,1536)
  const float* lng  = (const float*)d_in[12];
  const float* lnb  = (const float*)d_in[13];
  float* out = (float*)d_out;
  (void)in_sizes; (void)n_in; (void)out_size; (void)ws_size;

  char* p = (char*)d_ws;
  auto carve = [&](size_t bytes) -> void* {
    void* r = (void*)p; p += (bytes + 255) & ~(size_t)255; return r;
  };
  float* z     = (float*)carve((size_t)M_ * DFEAT_ * 4);          // residual stream
  float* xz    = (float*)carve((size_t)M_ * 2 * DINNER_ * 4);     // in_proj out
  float* u     = (float*)carve((size_t)M_ * DINNER_ * 4);
  float* delta = (float*)carve((size_t)M_ * DINNER_ * 4);
  float* dbl   = (float*)carve((size_t)M_ * DBLW_ * 4);
  unsigned short* abf  = (unsigned short*)carve((size_t)M_ * DINNER_ * 2);
  unsigned short* dtbf = (unsigned short*)carve((size_t)M_ * 64 * 2);
  unsigned short* wbf  = (unsigned short*)carve((size_t)2 * DINNER_ * DFEAT_ * 2);

  const int TB = 256;
  auto cvt = [&](const float* src, unsigned short* dst, long long R, int C, int Cp, int S, int off) {
    long long tot = R * Cp;
    cvt_pad_bf16<<<cdiv_ll(tot, TB), TB, 0, stream>>>(src, dst, C, Cp, S, off, tot);
  };
  auto gemm = [&](const unsigned short* A, const unsigned short* Wt, const float* bias,
                  float* Dst, int Mr, int Nr, int Kr, int act, int accum) {
    int ntiles = Nr / 16;
    int ng4    = ntiles / 4;
    int mb     = Mr / 16 / 8;                 // 64 macro-rows of 128
    if (ng4 > 0)
      gemm_bf16_wmma<4><<<mb * ng4, TB, 0, stream>>>(A, Wt, bias, Dst,
                                                     Mr, Nr, Kr, 0, ng4, act, accum);
    int rem = ntiles - ng4 * 4;
    if (rem > 0)
      gemm_bf16_wmma<1><<<mb * rem, TB, 0, stream>>>(A, Wt, bias, Dst,
                                                     Mr, Nr, Kr, ng4 * 4, rem, act, accum);
  };

  // ---- input projection: z = x @ piw^T + pib -------------------------------
  cvt(x,   abf, M_,     DIN_, DIN_, DIN_, 0);
  cvt(piw, wbf, DFEAT_, DIN_, DIN_, DIN_, 0);
  gemm(abf, wbf, pib, z, M_, DFEAT_, DIN_, 0, 0);

  for (int l = 0; l < NLAYERS_; ++l) {
    const float* inw_l = inw  + (size_t)l * 2 * DINNER_ * DFEAT_;
    const float* cw_l  = cw   + (size_t)l * DINNER_ * DCONV_;
    const float* cb_l  = cb   + (size_t)l * DINNER_;
    const float* xpw_l = xpw  + (size_t)l * DBLW_ * DINNER_;
    const float* dtw_l = dtw  + (size_t)l * DINNER_ * DTRANK_;
    const float* dtb_l = dtb  + (size_t)l * DINNER_;
    const float* al_l  = alog + (size_t)l * DINNER_ * DSTATE_;
    const float* Dv_l  = Dvp  + (size_t)l * DINNER_;
    const float* ow_l  = ow   + (size_t)l * DFEAT_ * DINNER_;

    // xz = z @ inw^T   (M x 3072, K=768)
    cvt(z,     abf, M_,          DFEAT_, DFEAT_, DFEAT_, 0);
    cvt(inw_l, wbf, 2 * DINNER_, DFEAT_, DFEAT_, DFEAT_, 0);
    gemm(abf, wbf, nullptr, xz, M_, 2 * DINNER_, DFEAT_, 0, 0);

    // u = silu(depthwise_conv(xp) + cb)  (also staged to bf16 in abf)
    conv_silu<<<cdiv_ll((long long)M_ * DINNER_, TB), TB, 0, stream>>>(xz, cw_l, cb_l, u, abf);

    // dbl = u @ xpw^T  (M x 80, K=1536)
    cvt(xpw_l, wbf, DBLW_, DINNER_, DINNER_, DINNER_, 0);
    gemm(abf, wbf, nullptr, dbl, M_, DBLW_, DINNER_, 0, 0);

    // delta = softplus(dt_lo @ dtw^T + dtb)  (K padded 48 -> 64)
    cvt(dbl,   dtbf, M_,      DTRANK_, 64, DBLW_,   0);
    cvt(dtw_l, wbf,  DINNER_, DTRANK_, 64, DTRANK_, 0);
    gemm(dtbf, wbf, dtb_l, delta, M_, DINNER_, 64, 1, 0);

    // sequential selective scan; produces gated bf16 activations in abf
    mamba_scan<<<cdiv_ll(B_ * DINNER_, TB), TB, 0, stream>>>(delta, u, dbl, xz,
                                                             al_l, Dv_l, abf);

    // z += p @ ow^T   (M x 768, K=1536), residual accumulate
    cvt(ow_l, wbf, DFEAT_, DINNER_, DINNER_, DINNER_, 0);
    gemm(abf, wbf, nullptr, z, M_, DFEAT_, DINNER_, 0, 1);
  }

  // final LayerNorm
  layernorm<<<cdiv_ll((long long)M_ * 32, TB), TB, 0, stream>>>(z, lng, lnb, out);
}
